// Multi_Order_CAM_Module_77653008711849
// MI455X (gfx1250) — matrix operand order repaired and checked by reading, not compiled, not run on hardware
//
#include <hip/hip_runtime.h>

// ---------------------------------------------------------------------------
// Multi-order channel-attention module, MI455X (gfx1250, wave32, WMMA bf16)
//
//   energy = Xf Xf^T            (bf16 WMMA, fp32 accum)
//   sigma  = (energy - s_c s_d / N) * 1/(N sqrt(N))   [derived, no 2nd gram]
//   attn   = beta*softmax(sigma) + (1-beta)*softmax(rowmax - energy)
//   out    = gamma * (attn Xf) + x                    (bf16 WMMA, fp32 accum)
//
// v2: added Xf^T staging (LDS-tiled transpose) so GEMM2's B operand is
//     loaded with contiguous b128 loads instead of 16 strided u16 loads.
// ---------------------------------------------------------------------------

typedef __bf16 bf16_t;
typedef bf16_t v16bf __attribute__((ext_vector_type(16)));
typedef bf16_t v8bf  __attribute__((ext_vector_type(8)));
typedef bf16_t v4bf  __attribute__((ext_vector_type(4)));
typedef float  v8f   __attribute__((ext_vector_type(8)));

#define BATCH 16
#define CCH   512      // channels
#define NSP   1024     // H*W
#define SCALE (1.0f / 32768.0f)   // 1/(N*sqrt(N)) = 1/(1024*32)

// ---------------------------------------------------------------------------
// WMMA wrapper: D = A(16x32 bf16) * B(32x16 bf16) + C(16x16 f32)
// ---------------------------------------------------------------------------
__device__ __forceinline__ v8f wmma_bf16(v16bf a, v16bf b, v8f c) {
  return __builtin_amdgcn_wmma_f32_16x16x32_bf16(
      /*neg_a=*/false, a, /*neg_b=*/false, b,
      /*c_mod=*/(short)0, c, /*reuse_a=*/false, /*reuse_b=*/false);
}

// A operand, 16x32 (MxK) bf16, row-major source with row stride `stride`.
// ISA 7.12.2: lane L holds row M=L%16; elements 0..7 = K[half*8 .. +7],
// elements 8..15 = K[16+half*8 .. +7], half = L/16.
__device__ __forceinline__ v16bf load_a_tile(const bf16_t* base, int stride,
                                             int r0, int k0, int lane) {
  const int row  = r0 + (lane & 15);
  const int half = lane >> 4;
  const bf16_t* p = base + row * stride + k0 + half * 8;
  v8bf lo = *(const v8bf*)(p);
  v8bf hi = *(const v8bf*)(p + 16);
  v16bf a;
#pragma unroll
  for (int i = 0; i < 8; ++i) { a[i] = lo[i]; a[i + 8] = hi[i]; }
  return a;
}

// B operand, 32x16 (KxN) bf16, where B[k][n] = src[(n0+n)*stride + k0+k]
// (row-block of a row-major matrix: used for X*X^T and for XfT in GEMM2).
// ISA 7.12.2 (B layout): lane L holds column N=L%16; elements 0..7 = K[half*8 .. +7],
// elements 8..15 = K[16+half*8 .. +7], half = L/16.
__device__ __forceinline__ v16bf load_b_rows(const bf16_t* base, int stride,
                                             int n0, int k0, int lane) {
  const int row = n0 + (lane & 15);
  const int kk  = k0 + ((lane >> 4) << 3);
  const bf16_t* p = base + row * stride + kk;
  v8bf lo = *(const v8bf*)(p);
  v8bf hi = *(const v8bf*)(p + 16);
  v16bf b;
#pragma unroll
  for (int i = 0; i < 8; ++i) { b[i] = lo[i]; b[i + 8] = hi[i]; }
  return b;
}

// C/D layout (ISA 7.12.2): VGPR r <-> row M = r + 8*(lane/16), col N = lane%16.
__device__ __forceinline__ void store_tile_f32(float* out, int stride,
                                               int r0, int c0, int lane, v8f acc) {
  const int m0 = (lane >> 4) * 8;
  const int n  = lane & 15;
#pragma unroll
  for (int r = 0; r < 8; ++r)
    out[(r0 + m0 + r) * stride + c0 + n] = acc[r];
}

// ---------------------------------------------------------------------------
// Kernel 1: fp32 -> bf16 convert + per-(b,c) row sums. One block per row.
// ---------------------------------------------------------------------------
__global__ void k_convert_rowsum(const float* __restrict__ x,
                                 bf16_t* __restrict__ xbf,
                                 float* __restrict__ rowsum) {
  __shared__ float red[256];
  const int row  = blockIdx.x;          // 0 .. B*C-1
  const int t    = threadIdx.x;         // 256 threads, 4 elems each
  const float* src = x + (size_t)row * NSP + t * 4;
  bf16_t*      dst = xbf + (size_t)row * NSP + t * 4;

  float4 v = *(const float4*)src;
  v4bf b;
  b[0] = (bf16_t)v.x; b[1] = (bf16_t)v.y; b[2] = (bf16_t)v.z; b[3] = (bf16_t)v.w;
  *(v4bf*)dst = b;

  red[t] = v.x + v.y + v.z + v.w;
  __syncthreads();
#pragma unroll
  for (int s = 128; s > 0; s >>= 1) {
    if (t < s) red[t] += red[t + s];
    __syncthreads();
  }
  if (t == 0) rowsum[row] = red[0];
}

// ---------------------------------------------------------------------------
// Kernel 1b: LDS-tiled transpose xbf (C x N) -> xbfT (N x C) per batch,
// 32x32 tiles, 256 threads (each moves 4 elements per phase).
// ---------------------------------------------------------------------------
__global__ void k_transpose(const bf16_t* __restrict__ xbf,
                            bf16_t* __restrict__ xbfT) {
  __shared__ bf16_t tile[32][33];       // +1 pad: no bank conflicts
  const int bid   = blockIdx.x;
  const int batch = bid >> 9;           // 512 tiles/batch: 16 (C) x 32 (N)
  const int t     = bid & 511;
  const int c0    = (t >> 5) * 32;      // channel tile base
  const int n0    = (t & 31) * 32;      // spatial tile base
  const bf16_t* src = xbf  + (size_t)batch * CCH * NSP;
  bf16_t*       dst = xbfT + (size_t)batch * NSP * CCH;
  const int tx = threadIdx.x & 31;
  const int ty = threadIdx.x >> 5;      // 8 rows per phase, 4 phases

#pragma unroll
  for (int p = 0; p < 4; ++p) {
    const int c = ty + p * 8;
    tile[c][tx] = src[(size_t)(c0 + c) * NSP + n0 + tx];
  }
  __syncthreads();
#pragma unroll
  for (int p = 0; p < 4; ++p) {
    const int n = ty + p * 8;
    dst[(size_t)(n0 + n) * CCH + c0 + tx] = tile[tx][n];
  }
}

// ---------------------------------------------------------------------------
// Kernel 2: energy = Xf * Xf^T (per batch). One wave per 32x32 output tile:
// 2x2 accumulators, 4 WMMAs per 32-wide K step, K = 1024 -> 128 v_wmma/wave.
// ---------------------------------------------------------------------------
__global__ void k_gram(const bf16_t* __restrict__ xbf,
                       float* __restrict__ energy) {
  const int wid   = (blockIdx.x * blockDim.x + threadIdx.x) >> 5;
  const int lane  = threadIdx.x & 31;
  const int batch = wid >> 8;             // 256 tiles per batch (16x16)
  const int t     = wid & 255;
  const int ti    = (t >> 4) * 32;        // row tile
  const int tj    = (t & 15) * 32;        // col tile
  const bf16_t* Xb = xbf + (size_t)batch * CCH * NSP;

  v8f acc00 = {0,0,0,0,0,0,0,0}, acc01 = {0,0,0,0,0,0,0,0};
  v8f acc10 = {0,0,0,0,0,0,0,0}, acc11 = {0,0,0,0,0,0,0,0};

  for (int k0 = 0; k0 < NSP; k0 += 32) {
    if (k0 + 32 < NSP)   // global_prefetch_b8 for the next K block
      __builtin_prefetch(Xb + (ti + (lane & 15)) * NSP + k0 + 32, 0, 1);
    v16bf a0 = load_a_tile(Xb, NSP, ti,      k0, lane);
    v16bf a1 = load_a_tile(Xb, NSP, ti + 16, k0, lane);
    v16bf b0 = load_b_rows(Xb, NSP, tj,      k0, lane);
    v16bf b1 = load_b_rows(Xb, NSP, tj + 16, k0, lane);
    acc00 = wmma_bf16(a0, b0, acc00);
    acc01 = wmma_bf16(a0, b1, acc01);
    acc10 = wmma_bf16(a1, b0, acc10);
    acc11 = wmma_bf16(a1, b1, acc11);
  }

  float* Eb = energy + (size_t)batch * CCH * CCH;
  store_tile_f32(Eb, CCH, ti,      tj,      lane, acc00);
  store_tile_f32(Eb, CCH, ti,      tj + 16, lane, acc01);
  store_tile_f32(Eb, CCH, ti + 16, tj,      lane, acc10);
  store_tile_f32(Eb, CCH, ti + 16, tj + 16, lane, acc11);
}

// ---------------------------------------------------------------------------
// Kernel 3: per row c of energy: blended dual softmax -> attention (bf16).
//   sigma_d           = (e_d - s_c*s_d/N) * SCALE
//   softmax(max-e)_d  = exp(emin - e_d) / sum
// One 256-thread block per (b,c) row of 512 values (2 per thread).
// ---------------------------------------------------------------------------
__global__ void k_softmax(const float* __restrict__ energy,
                          const float* __restrict__ rowsum,
                          const float* __restrict__ beta,
                          bf16_t* __restrict__ attn) {
  __shared__ float red[256];
  const int r     = blockIdx.x;     // 0 .. B*C-1
  const int batch = r / CCH;
  const int c     = r % CCH;
  const int t     = threadIdx.x;

  const float* erow = energy + (size_t)r * CCH;
  const float* srow = rowsum + (size_t)batch * CCH;
  const float  sc   = srow[c];

  const float e0 = erow[t], e1 = erow[t + 256];
  const float sg0 = (e0 - sc * srow[t]       * (1.0f / NSP)) * SCALE;
  const float sg1 = (e1 - sc * srow[t + 256] * (1.0f / NSP)) * SCALE;

  // row min of energy
  red[t] = fminf(e0, e1);
  __syncthreads();
#pragma unroll
  for (int s = 128; s > 0; s >>= 1) { if (t < s) red[t] = fminf(red[t], red[t + s]); __syncthreads(); }
  const float emin = red[0];
  __syncthreads();

  // row max of sigma
  red[t] = fmaxf(sg0, sg1);
  __syncthreads();
#pragma unroll
  for (int s = 128; s > 0; s >>= 1) { if (t < s) red[t] = fmaxf(red[t], red[t + s]); __syncthreads(); }
  const float smax = red[0];
  __syncthreads();

  const float ws0 = __expf(sg0 - smax),  ws1 = __expf(sg1 - smax);
  const float we0 = __expf(emin - e0),   we1 = __expf(emin - e1);

  // sum of sigma-exps
  red[t] = ws0 + ws1;
  __syncthreads();
#pragma unroll
  for (int s = 128; s > 0; s >>= 1) { if (t < s) red[t] += red[t + s]; __syncthreads(); }
  const float sum_s = red[0];
  __syncthreads();

  // sum of energy-exps
  red[t] = we0 + we1;
  __syncthreads();
#pragma unroll
  for (int s = 128; s > 0; s >>= 1) { if (t < s) red[t] += red[t + s]; __syncthreads(); }
  const float sum_e = red[0];

  const float b  = beta[0];
  const float is = b / sum_s, ie = (1.0f - b) / sum_e;
  bf16_t* arow = attn + (size_t)r * CCH;
  arow[t]       = (bf16_t)(ws0 * is + we0 * ie);
  arow[t + 256] = (bf16_t)(ws1 * is + we1 * ie);
}

// ---------------------------------------------------------------------------
// Kernel 4: out = gamma * (attn @ Xf) + x. One wave per 32x32 output tile,
// K = C = 512 -> 16 K-steps * 4 = 64 v_wmma/wave; fused epilogue.
// B operand read from XfT (N x C) -> contiguous b128 loads.
// ---------------------------------------------------------------------------
__global__ void k_attn_apply(const bf16_t* __restrict__ attn,
                             const bf16_t* __restrict__ xbfT,
                             const float* __restrict__ x,
                             const float* __restrict__ gamma,
                             float* __restrict__ out) {
  const int wid   = (blockIdx.x * blockDim.x + threadIdx.x) >> 5;
  const int lane  = threadIdx.x & 31;
  const int batch = wid / 512;            // 512 tiles per batch (16 x 32)
  const int t     = wid % 512;
  const int ti    = (t >> 5) * 32;        // channel tile (M)
  const int tj    = (t & 31) * 32;        // spatial tile (N)
  const bf16_t* Ab = attn + (size_t)batch * CCH * CCH;   // 512x512 row-major
  const bf16_t* Tb = xbfT + (size_t)batch * NSP * CCH;   // 1024x512 row-major

  v8f acc00 = {0,0,0,0,0,0,0,0}, acc01 = {0,0,0,0,0,0,0,0};
  v8f acc10 = {0,0,0,0,0,0,0,0}, acc11 = {0,0,0,0,0,0,0,0};

  for (int k0 = 0; k0 < CCH; k0 += 32) {
    if (k0 + 32 < CCH)   // global_prefetch_b8 for the next K block
      __builtin_prefetch(Ab + (ti + (lane & 15)) * CCH + k0 + 32, 0, 1);
    v16bf a0 = load_a_tile(Ab, CCH, ti,      k0, lane);
    v16bf a1 = load_a_tile(Ab, CCH, ti + 16, k0, lane);
    // B[k][n] = Xf[k][n] = XfT[n][k]: row-block loads of XfT
    v16bf b0 = load_b_rows(Tb, CCH, tj,      k0, lane);
    v16bf b1 = load_b_rows(Tb, CCH, tj + 16, k0, lane);
    acc00 = wmma_bf16(a0, b0, acc00);
    acc01 = wmma_bf16(a0, b1, acc01);
    acc10 = wmma_bf16(a1, b0, acc10);
    acc11 = wmma_bf16(a1, b1, acc11);
  }

  const float g  = gamma[0];
  const int m0   = (lane >> 4) * 8;
  const int n    = lane & 15;
  const int base = batch * CCH;
#pragma unroll
  for (int jj = 0; jj < 2; ++jj) {
#pragma unroll
    for (int ii = 0; ii < 2; ++ii) {
      v8f acc = (ii == 0) ? (jj == 0 ? acc00 : acc01)
                          : (jj == 0 ? acc10 : acc11);
#pragma unroll
      for (int r = 0; r < 8; ++r) {
        const size_t idx = (size_t)(base + ti + ii * 16 + m0 + r) * NSP
                         + tj + jj * 16 + n;
        out[idx] = g * acc[r] + x[idx];
      }
    }
  }
}

// ---------------------------------------------------------------------------
// Launch
// ---------------------------------------------------------------------------
extern "C" void kernel_launch(void* const* d_in, const int* in_sizes, int n_in,
                              void* d_out, int out_size, void* d_ws, size_t ws_size,
                              hipStream_t stream) {
  const float* x     = (const float*)d_in[0];   // (16,512,32,32)
  const float* gamma = (const float*)d_in[1];   // (1,)
  const float* beta  = (const float*)d_in[2];   // (1,)
  float*       out   = (float*)d_out;           // (16,512,32,32)

  char* ws = (char*)d_ws;
  const size_t xbf_bytes  = (size_t)BATCH * CCH * NSP * sizeof(bf16_t); // 16.8 MB
  const size_t xbT_bytes  = (size_t)BATCH * NSP * CCH * sizeof(bf16_t); // 16.8 MB
  const size_t en_bytes   = (size_t)BATCH * CCH * CCH * sizeof(float);  // 16.8 MB
  const size_t attn_bytes = (size_t)BATCH * CCH * CCH * sizeof(bf16_t); //  8.4 MB
  bf16_t* xbf    = (bf16_t*)(ws);
  bf16_t* xbfT   = (bf16_t*)(ws + xbf_bytes);
  float*  energy = (float*) (ws + xbf_bytes + xbT_bytes);
  bf16_t* attn   = (bf16_t*)(ws + xbf_bytes + xbT_bytes + en_bytes);
  float*  rowsum = (float*) (ws + xbf_bytes + xbT_bytes + en_bytes + attn_bytes);
  (void)in_sizes; (void)n_in; (void)out_size; (void)ws_size;

  // 1) convert + row sums: one block per (b,c) row
  k_convert_rowsum<<<BATCH * CCH, 256, 0, stream>>>(x, xbf, rowsum);

  // 1b) transpose to XfT for GEMM2's B operand
  k_transpose<<<BATCH * 512, 256, 0, stream>>>(xbf, xbfT);

  // 2) gram: 16 batches * 256 tiles, 8 waves per 256-thread block
  k_gram<<<(BATCH * 256) / 8, 256, 0, stream>>>(xbf, energy);

  // 3) dual softmax blend: one block per (b,c) row
  k_softmax<<<BATCH * CCH, 256, 0, stream>>>(energy, rowsum, beta, attn);

  // 4) attn @ XfT^T + epilogue: 16 batches * 512 tiles, 8 waves per block
  k_attn_apply<<<(BATCH * 512) / 8, 256, 0, stream>>>(attn, xbfT, x, gamma, out);
}
